// AttentionModel_39848706572604
// MI455X (gfx1250) — matrix operand order repaired and checked by reading, not compiled, not run on hardware
//
#include <hip/hip_runtime.h>
#include <stdint.h>

// ---------------------------------------------------------------------------
// Attention forward for MI455X (gfx1250, wave32, WMMA + TDM).
//   q = x Wq^T + bq ; k = x Wk^T + bk ; v = x Wv^T + bv        (f16 in d_ws)
//   out = softmax(q k^T / sqrt(E)) v                            (f32 to d_out)
// B=4, S=2048, E=1024.  Workspace use: 3 * 16 MB = 48 MB (f16 QKV).
// ---------------------------------------------------------------------------

typedef _Float16 half_t;
typedef __attribute__((ext_vector_type(16))) _Float16     v16h;
typedef __attribute__((ext_vector_type(8)))  _Float16     v8h;
typedef __attribute__((ext_vector_type(8)))  float        v8f;
typedef __attribute__((ext_vector_type(4)))  unsigned int v4u;
typedef __attribute__((ext_vector_type(8)))  int          v8i;
typedef __attribute__((ext_vector_type(4)))  int          v4i;

constexpr int Bn = 4;
constexpr int Sn = 2048;
constexpr int En = 1024;
constexpr int SSTR = Sn + 1;   // padded f32 row stride for scores (bank spread)

// ---------------------------------------------------------------------------
// TDM: issue a tensor_load_to_lds of a 32x1024 f16 tile (one V key-tile).
// D# layout per CDNA5 ISA ch.8: group0 = {flags, lds_addr, global_addr, type},
// group1 = {data_size, tensor dims 1024x2048, tile 1024x32, stride 1024}.
// Tracked with TENSORcnt (in-order per wave).
// ---------------------------------------------------------------------------
__device__ static inline void tdm_load_v_tile(const half_t* gsrc, unsigned lds_off)
{
    const unsigned long long ga = (unsigned long long)(uintptr_t)gsrc;
    v4u g0;
    g0[0] = 1u;                                      // count=1, user descriptor
    g0[1] = lds_off;                                 // lds_addr (bytes)
    g0[2] = (unsigned)(ga & 0xFFFFFFFFu);            // global_addr[31:0]
    g0[3] = (unsigned)((ga >> 32) & 0x01FFFFFFu)     // global_addr[56:32]
          | (2u << 30);                              // type = 2 (image)
    v8i g1;
    g1[0] = (int)(1u << 16);                         // data_size=1 (2B), no pad
    g1[1] = (int)(1024u << 16);                      // tensor_dim0 lo16 in [63:48]
    g1[2] = (int)((1024u >> 16) | (2048u << 16));    // dim0 hi16 | tensor_dim1 lo16
    g1[3] = (int)((2048u >> 16) | (1024u << 16));    // dim1 hi16 | tile_dim0=1024
    g1[4] = 32;                                      // tile_dim1=32, tile_dim2=0
    g1[5] = 1024;                                    // tensor_dim0_stride lo32
    g1[6] = 0;                                       // stride0 hi | stride1 lo
    g1[7] = 0;                                       // stride1 hi
    v4i g2 = {0, 0, 0, 0};                           // dims 2/3 unused (2D tile)
    v4i g3 = {0, 0, 0, 0};
#if defined(__clang_major__) && (__clang_major__ >= 23)
    v8i g1b = {0, 0, 0, 0, 0, 0, 0, 0};
    __builtin_amdgcn_tensor_load_to_lds(g0, g1, g2, g3, g1b, 0);
#else
    __builtin_amdgcn_tensor_load_to_lds(g0, g1, g2, g3, 0);
#endif
}

// ---------------------------------------------------------------------------
// Kernel 1: fused QKV projection.  grid = (M/16, E/512, 3), block = 256 (8 waves)
// Each wave computes a 16x64 tile of one of Q/K/V (selected by blockIdx.z).
// A (16x32 f16) = x tile (rows = flattened b*s), B (32x16 f16) = W^T tile:
// column n of B is row (nBase+n) of W, contiguous in e -> coalesced loads.
// ---------------------------------------------------------------------------
__global__ __launch_bounds__(256) void qkv_proj_kernel(
    const float* __restrict__ x,
    const float* __restrict__ Wq, const float* __restrict__ bq,
    const float* __restrict__ Wk, const float* __restrict__ bk,
    const float* __restrict__ Wv, const float* __restrict__ bv,
    half_t* __restrict__ Qh, half_t* __restrict__ Kh, half_t* __restrict__ Vh)
{
    const float* W;
    const float* bias;
    half_t* dst;
    if (blockIdx.z == 0)      { W = Wq; bias = bq; dst = Qh; }
    else if (blockIdx.z == 1) { W = Wk; bias = bk; dst = Kh; }
    else                      { W = Wv; bias = bv; dst = Vh; }

    const int lane = threadIdx.x & 31;
    const int wave = threadIdx.x >> 5;
    const int l16  = lane & 15;
    const int hi   = lane >> 4;          // 0: lanes 0-15, 1: lanes 16-31
    const int mBase = blockIdx.x * 16;                // row tile (b*s flattened)
    const int nBase = blockIdx.y * 512 + wave * 64;   // output-feature tile

    // A-fragment K offsets: lanes 0-15 hold K 0..15,
    // lanes 16-31 hold K 16..31.  B-fragment: lanes 0-15 K 0..15,
    // lanes 16-31 K 16..31, one column per lane.
    const int aoff = hi * 16;
    const int boff = hi * 16;

    v8f acc[4] = {};
    const float* xRow = x + (size_t)(mBase + l16) * En;

    for (int kk = 0; kk < En; kk += 32) {
        v16h a;
#pragma unroll
        for (int j = 0; j < 8; ++j) {
            a[j]     = (half_t)xRow[kk + aoff + j];
            a[8 + j] = (half_t)xRow[kk + 8 + aoff + j];
        }
#pragma unroll
        for (int t = 0; t < 4; ++t) {
            const float* wRow = W + (size_t)(nBase + t * 16 + l16) * En + kk + boff;
            v16h bm;
#pragma unroll
            for (int i = 0; i < 16; ++i) bm[i] = (half_t)wRow[i];
            acc[t] = __builtin_amdgcn_wmma_f32_16x16x32_f16(
                false, a, false, bm, (short)0, acc[t], false, false);
        }
    }

    // Epilogue: add bias, convert to f16, store.  D layout: lane = col (n%16),
    // vgpr r = row r + 8*hi.
#pragma unroll
    for (int t = 0; t < 4; ++t) {
        const int col = nBase + t * 16 + l16;
        const float bvv = bias[col];
#pragma unroll
        for (int r = 0; r < 8; ++r) {
            const int row = mBase + r + hi * 8;
            dst[(size_t)row * En + col] = (half_t)(acc[t][r] + bvv);
        }
    }
}

// ---------------------------------------------------------------------------
// Kernel 2: fused attention for one (batch, 16-query-row) strip.
// grid = B*S/16 = 512 blocks, 256 threads (8 waves).
// LDS: scores f32 [16][2049] (128KB+pad) + Q f16 [16][1024] (32KB)
//      + V double-buffer f16 2x[32][1024] (128KB) + rowSum[16]  = ~288 KB.
// Phase 1: S = Q K^T / 32  (WMMA, each wave does 256 key columns)
// Phase 2: row softmax (unnormalized exp + row sums, wave32 shfl reductions)
// Phase 3: O = P V (WMMA); V tiles streamed by the Tensor Data Mover into a
//          double buffer (TENSORcnt-paced), overlapping DMA with WMMA.
// ---------------------------------------------------------------------------
__global__ __launch_bounds__(256) void flash_attn_kernel(
    const half_t* __restrict__ Qh, const half_t* __restrict__ Kh,
    const half_t* __restrict__ Vh, float* __restrict__ out)
{
    extern __shared__ char smem[];
    float*  sS     = (float*)smem;                                  // [16][SSTR]
    half_t* sQ     = (half_t*)(smem + 16 * SSTR * 4);               // [16][1024]
    half_t* sV     = (half_t*)(smem + 16 * SSTR * 4 + 16 * En * 2); // 2x[32][1024]
    float*  rowSum = (float*)(smem + 16 * SSTR * 4 + 16 * En * 2 + 2 * 32 * En * 2);

    const int b     = blockIdx.x >> 7;          // S/16 = 128 strips per batch
    const int qBase = (blockIdx.x & 127) * 16;
    const int lane  = threadIdx.x & 31;
    const int wave  = threadIdx.x >> 5;
    const int l16   = lane & 15;
    const int hi    = lane >> 4;
    const int aoff  = hi * 16;
    const int boff  = hi * 16;

    // ---- Phase 0: stage Q strip (16 x 1024 f16) into LDS --------------------
    {
        const uint4* src = (const uint4*)(Qh + (size_t)(b * Sn + qBase) * En);
        uint4* dst4 = (uint4*)sQ;
        for (int i = threadIdx.x; i < 16 * 128; i += 256) dst4[i] = src[i];
    }
    // Prologue for phase 3: TDM-load V tile 0 into buffer 0 (overlaps phase 1/2)
    const half_t* Vbase = Vh + (size_t)b * Sn * En;
    const unsigned sV0off = (unsigned)(uintptr_t)sV;
    if (wave == 0) tdm_load_v_tile(Vbase, sV0off);
    __syncthreads();

    // ---- Phase 1: scores = Q K^T * (1/sqrt(E)) into LDS ---------------------
    {
        const float scale = 0.03125f;  // 1/32
        for (int nt = 0; nt < 16; ++nt) {
            const int n = wave * 256 + nt * 16 + l16;          // key index
            const half_t* kRow = Kh + (size_t)(b * Sn + n) * En;
            __builtin_prefetch(kRow + 16 * En, 0, 1);          // next key row
            v8f c = {};
            for (int kk = 0; kk < En; kk += 32) {
                v8h lo  = *(const v8h*)(sQ + l16 * En + kk + aoff);
                v8h hiv = *(const v8h*)(sQ + l16 * En + kk + 8 + aoff);
                v16h a;
#pragma unroll
                for (int j = 0; j < 8; ++j) { a[j] = lo[j]; a[8 + j] = hiv[j]; }
                v16h bm = *(const v16h*)(kRow + kk + boff);    // 32B contiguous
                c = __builtin_amdgcn_wmma_f32_16x16x32_f16(
                    false, a, false, bm, (short)0, c, false, false);
            }
#pragma unroll
            for (int r = 0; r < 8; ++r)
                sS[(r + hi * 8) * SSTR + n] = c[r] * scale;
        }
    }
    __syncthreads();

    // ---- Phase 2: softmax stats; store exp(s - max), keep row sums ----------
    {
        for (int rr = 0; rr < 2; ++rr) {
            const int row = wave * 2 + rr;
            float m = -3.0e38f;
            for (int j = 0; j < 64; ++j)
                m = fmaxf(m, sS[row * SSTR + lane + 32 * j]);
#pragma unroll
            for (int off = 16; off; off >>= 1)
                m = fmaxf(m, __shfl_xor(m, off, 32));
            float ssum = 0.f;
            for (int j = 0; j < 64; ++j) {
                float v = __expf(sS[row * SSTR + lane + 32 * j] - m);
                sS[row * SSTR + lane + 32 * j] = v;
                ssum += v;
            }
#pragma unroll
            for (int off = 16; off; off >>= 1)
                ssum += __shfl_xor(ssum, off, 32);
            if (lane == 0) rowSum[row] = ssum;
        }
    }
    __syncthreads();

    // ---- Phase 3: O = P V, wave owns output columns [wave*128, wave*128+128)
    // V tiles streamed by TDM into a double buffer; s_wait_tensorcnt(1) at the
    // top of each iteration guarantees the current buffer landed (in-order),
    // while the next tile's DMA runs concurrently with the WMMAs.
    v8f acc[8] = {};
    for (int kt = 0; kt < Sn; kt += 32) {
        const int ib = (kt >> 5) & 1;
        const half_t* sVc = sV + ib * 32 * En;
        if (wave == 0) {
            const int ktn = (kt + 32) & (Sn - 1);   // wrap: last issue is unused
            tdm_load_v_tile(Vbase + (size_t)ktn * En,
                            sV0off + (unsigned)((ib ^ 1) * 32 * En * 2));
            __builtin_amdgcn_s_wait_tensorcnt(1);
        }
        __syncthreads();

        // A = P tile (16 x 32), f32 LDS -> f16 fragment
        v16h a;
#pragma unroll
        for (int j = 0; j < 8; ++j) {
            a[j]     = (half_t)sS[l16 * SSTR + kt + aoff + j];
            a[8 + j] = (half_t)sS[l16 * SSTR + kt + 8 + aoff + j];
        }
#pragma unroll
        for (int nt = 0; nt < 8; ++nt) {
            const int e = wave * 128 + nt * 16 + l16;
            v16h bm;
#pragma unroll
            for (int i = 0; i < 16; ++i) bm[i] = sVc[(boff + i) * En + e];
            acc[nt] = __builtin_amdgcn_wmma_f32_16x16x32_f16(
                false, a, false, bm, (short)0, acc[nt], false, false);
        }
        __syncthreads();
    }

    // ---- Epilogue: normalize by row sums, store f32 -------------------------
#pragma unroll
    for (int nt = 0; nt < 8; ++nt) {
        const int e = wave * 128 + nt * 16 + l16;
#pragma unroll
        for (int r = 0; r < 8; ++r) {
            const int row = r + hi * 8;
            const float inv = 1.0f / rowSum[row];
            out[(size_t)(b * Sn + qBase + row) * En + e] = acc[nt][r] * inv;
        }
    }
}

// ---------------------------------------------------------------------------
extern "C" void kernel_launch(void* const* d_in, const int* in_sizes, int n_in,
                              void* d_out, int out_size, void* d_ws, size_t ws_size,
                              hipStream_t stream)
{
    const float* x  = (const float*)d_in[0];
    const float* Wq = (const float*)d_in[1];
    const float* bq = (const float*)d_in[2];
    const float* Wk = (const float*)d_in[3];
    const float* bk = (const float*)d_in[4];
    const float* Wv = (const float*)d_in[5];
    const float* bv = (const float*)d_in[6];
    float* out = (float*)d_out;

    // Workspace: f16 Q, K, V  (16 MB each, 48 MB total)
    half_t* Qh = (half_t*)d_ws;
    half_t* Kh = Qh + (size_t)Bn * Sn * En;
    half_t* Vh = Kh + (size_t)Bn * Sn * En;

    dim3 g1(Bn * Sn / 16, En / 512, 3);
    qkv_proj_kernel<<<g1, 256, 0, stream>>>(x, Wq, bq, Wk, bk, Wv, bv, Qh, Kh, Vh);

    const size_t smem = (size_t)16 * SSTR * 4      // scores f32 (padded rows)
                      + (size_t)16 * En * 2        // Q f16
                      + (size_t)2 * 32 * En * 2    // V tile double buffer f16
                      + 64;                        // row sums
    flash_attn_kernel<<<dim3(Bn * Sn / 16), 256, smem, stream>>>(Qh, Kh, Vh, out);
}
